// GPS_80152679678750
// MI455X (gfx1250) — matrix operand order repaired and checked by reading, not compiled, not run on hardware
//
#include <hip/hip_runtime.h>
#include <math.h>

// ---- problem constants ----
#define NN    16384
#define EE    262144
#define BB    64
#define GG    256
#define CH_   256
#define HEADS_ 8
#define LL    4
#define PED   20
#define PEO   25
#define IND   68

typedef __attribute__((ext_vector_type(16))) __bf16          v16bf;
typedef __attribute__((ext_vector_type(16))) unsigned short  v16us;
typedef __attribute__((ext_vector_type(8)))  float           v8f;

__device__ __forceinline__ unsigned short f2bfu(float f) {
  return (unsigned short)(__builtin_bit_cast(unsigned int, f) >> 16);
}

// =====================================================================
// WMMA GEMM:  C[M x Nout] = act( A[M x K] @ W[Nout x K]^T + bias (+resid) )
// 256 threads / block = 8 waves in a 4x2 grid.
// Block tile: 128 x TN (TN = 128 or 64); wave tile 32 x (TN/2).
// Per wave per K-step: 2 A-frags x (TN/32) B-frags WMMAs, bf16 in, f32 acc.
// LDS double-buffered (one barrier per K-step); global prefetch of tile k+2.
// Requires: M % 128 == 0, Nout % TN == 0, K % 32 == 0.
// =====================================================================
template<int TN>
__global__ __launch_bounds__(256)
void gemm_bf16_wmma(const float* __restrict__ A, const float* __restrict__ W,
                    const float* __restrict__ bias, const float* __restrict__ resid,
                    float* __restrict__ C, long M, int K, int Nout, int act)
{
  constexpr int TM   = 128;
  constexpr int BF   = TN / 32;          // B-frags per wave (4 or 2)
  constexpr int EPTB = TN * 32 / 256;    // B-tile elements per thread (16 or 8)

  __shared__ unsigned short As[2][TM][32];
  __shared__ unsigned short Bs[2][TN][32];

  const int tid  = threadIdx.x;
  const int wave = tid >> 5, lane = tid & 31;
  const int half = lane >> 4, l16 = lane & 15;
  const int wrow = wave >> 1;            // 0..3 -> 32-row slab
  const int wcol = wave & 1;             // 0..1 -> (TN/2)-col slab
  const long tileM = (long)blockIdx.y * TM;
  const int  tileN = blockIdx.x * TN;

  v8f acc[2][BF];
#pragma unroll
  for (int a = 0; a < 2; ++a)
#pragma unroll
    for (int b = 0; b < BF; ++b)
      acc[a][b] = (v8f){0.f,0.f,0.f,0.f,0.f,0.f,0.f,0.f};

  auto stage = [&](int buf, int k0) {
#pragma unroll
    for (int j = 0; j < 16; ++j) {                 // A tile: 128x32, 16/thread
      const int idx = tid * 16 + j;
      const int r = idx >> 5, c = idx & 31;
      As[buf][r][c] = f2bfu(A[(tileM + r) * (long)K + k0 + c]);
    }
#pragma unroll
    for (int j = 0; j < EPTB; ++j) {               // B tile: TNx32
      const int idx = tid * EPTB + j;
      const int r = idx >> 5, c = idx & 31;
      Bs[buf][r][c] = f2bfu(W[(long)(tileN + r) * K + k0 + c]);
    }
  };

  stage(0, 0);
  int buf = 0;
  for (int k0 = 0; k0 < K; k0 += 32) {
    __syncthreads();
    if (k0 + 32 < K) {
      stage(buf ^ 1, k0 + 32);
      if (k0 + 64 < K) {                           // prefetch tile k+2
        __builtin_prefetch(&A[(tileM + (tid >> 1)) * (long)K + k0 + 64], 0, 1);
        __builtin_prefetch(&W[(long)(tileN + ((tid >> 1) & (TN - 1))) * K + k0 + 64], 0, 1);
      }
    }

    // B fragments (32x16, K x N): lane half selects K 0..15 / 16..31, col = l16
    v16bf bfr[BF];
    const int cw = wcol * (TN / 2);
#pragma unroll
    for (int b = 0; b < BF; ++b) {
      v16us bu;
#pragma unroll
      for (int i = 0; i < 16; ++i)
        bu[i] = Bs[buf][cw + b * 16 + l16][half * 16 + i];
      bfr[b] = __builtin_bit_cast(v16bf, bu);
    }
    // A fragments (16x32): lanes 0-15 K 0..15, lanes 16-31 K 16..31
#pragma unroll
    for (int a = 0; a < 2; ++a) {
      v16us au;
      const int r = wrow * 32 + a * 16 + l16;
#pragma unroll
      for (int i = 0; i < 8; ++i) {
        au[i]     = As[buf][r][half * 16 + i];
        au[i + 8] = As[buf][r][half * 16 + 8 + i];
      }
      const v16bf af = __builtin_bit_cast(v16bf, au);
#pragma unroll
      for (int b = 0; b < BF; ++b)
        acc[a][b] = __builtin_amdgcn_wmma_f32_16x16x32_bf16(
            false, af, false, bfr[b], (short)0, acc[a][b], false, false);
    }
    buf ^= 1;
  }

  // epilogue: C frag layout — col = l16, half selects rows r / r+8
#pragma unroll
  for (int a = 0; a < 2; ++a) {
    const long rbase = tileM + wrow * 32 + a * 16 + half * 8;
#pragma unroll
    for (int b = 0; b < BF; ++b) {
      const int col = tileN + wcol * (TN / 2) + b * 16 + l16;
      const float bi = bias[col];
#pragma unroll
      for (int r = 0; r < 8; ++r) {
        const long row = rbase + r;
        float v = acc[a][b][r] + bi;
        if (resid) v += resid[row * Nout + col];
        if (act) v = fmaxf(v, 0.f);
        C[row * (long)Nout + col] = v;
      }
    }
  }
}

// =====================================================================
// LayerNorm over rows of width 256; one wave per row (wave32).
// =====================================================================
__global__ __launch_bounds__(256)
void ln256(const float* __restrict__ in, const float* __restrict__ g,
           const float* __restrict__ b, float* __restrict__ out, long R, float eps)
{
  const int wave = threadIdx.x >> 5, lane = threadIdx.x & 31;
  const long row = (long)blockIdx.x * 8 + wave;
  if (row >= R) return;
  const float* p = in + row * 256;
  float v[8], s = 0.f;
#pragma unroll
  for (int j = 0; j < 8; ++j) { v[j] = p[lane + 32 * j]; s += v[j]; }
#pragma unroll
  for (int o = 16; o; o >>= 1) s += __shfl_xor(s, o, 32);
  const float mu = s * (1.f / 256.f);
  float q = 0.f;
#pragma unroll
  for (int j = 0; j < 8; ++j) { float d = v[j] - mu; q += d * d; }
#pragma unroll
  for (int o = 16; o; o >>= 1) q += __shfl_xor(q, o, 32);
  const float rs = rsqrtf(q * (1.f / 256.f) + eps);
  float* op = out + row * 256;
#pragma unroll
  for (int j = 0; j < 8; ++j) {
    const int c = lane + 32 * j;
    op[c] = (v[j] - mu) * rs * g[c] + b[c];
  }
}

// ===== BatchNorm: stats (block per channel) + apply (elementwise) =====
__global__ __launch_bounds__(256)
void bn_stats(const float* __restrict__ a, const float* __restrict__ b,
              float* __restrict__ mean, float* __restrict__ rstd, int R)
{
  const int c = blockIdx.x, tid = threadIdx.x;
  float s = 0.f, q = 0.f;
  for (int r = tid; r < R; r += 256) {
    float v = a[(long)r * 256 + c];
    if (b) v += b[(long)r * 256 + c];
    s += v; q += v * v;
  }
  __shared__ float ss[256], qq[256];
  ss[tid] = s; qq[tid] = q; __syncthreads();
  for (int o = 128; o; o >>= 1) {
    if (tid < o) { ss[tid] += ss[tid + o]; qq[tid] += qq[tid + o]; }
    __syncthreads();
  }
  if (tid == 0) {
    const float mu = ss[0] / (float)R;
    const float var = qq[0] / (float)R - mu * mu;
    mean[c] = mu; rstd[c] = rsqrtf(var + 1e-5f);
  }
}

__global__ __launch_bounds__(256)
void bn_apply(const float* __restrict__ a, const float* __restrict__ b,
              const float* __restrict__ g, const float* __restrict__ bet,
              const float* __restrict__ mean, const float* __restrict__ rstd,
              float* __restrict__ out)
{
  const long idx = (long)blockIdx.x * 256 + threadIdx.x;
  const int c = (int)(idx & 255);
  float v = a[idx];
  if (b) v += b[idx];
  out[idx] = (v - mean[c]) * rstd[c] * g[c] + bet[c];
}

// ===== elementwise helpers =====
__global__ __launch_bounds__(256)
void ew_add(const float* __restrict__ a, const float* __restrict__ b,
            float* __restrict__ out)
{
  const long i = (long)blockIdx.x * 256 + threadIdx.x;
  out[i] = a[i] + b[i];
}

__global__ __launch_bounds__(256)
void slice_xs(const float* __restrict__ t, float* __restrict__ xs)
{
  const long idx = (long)blockIdx.x * 256 + threadIdx.x;   // over NN*256
  const long i = idx >> 8; const int c = (int)(idx & 255);
  xs[idx] = t[i * 1024 + 768 + c];
}

__global__ __launch_bounds__(256)
void gate_gelu(const float* __restrict__ t, const float* __restrict__ xs2,
               float* __restrict__ filt)
{
  const long idx = (long)blockIdx.x * 256 + threadIdx.x;   // over NN*512
  const long i = idx >> 9; const int c = (int)(idx & 511);
  const float gv = t[i * 1024 + c];
  const float other = (c < 256) ? t[i * 1024 + 512 + c] : xs2[i * 256 + (c - 256)];
  const float gelu = 0.5f * gv * (1.f + erff(gv * 0.7071067811865476f));
  filt[idx] = gelu * other;
}

// ===== GINE message + scatter-add (GLOBAL_ATOMIC_ADD_F32) =====
__global__ __launch_bounds__(256)
void gine_msg(const float* __restrict__ xs, const float* __restrict__ ealin,
              const int* __restrict__ ei, float* __restrict__ aggr)
{
  const long e = blockIdx.x; const int c = threadIdx.x;
  const int src = ei[e], dst = ei[EE + e];
  const float v = fmaxf(xs[(long)src * 256 + c] + ealin[e * 256 + c], 0.f);
  atomicAdd(&aggr[(long)dst * 256 + c], v);
}

// ===== fused attention: one thread per (node, head), online softmax =====
__global__ __launch_bounds__(256)
void attn_online(const float* __restrict__ qkv, float* __restrict__ o)
{
  const int t = blockIdx.x * 256 + threadIdx.x;   // NN*HEADS threads
  const int h = t & 7;
  const long node = t >> 3;
  const long base = (node >> 8) << 8;             // batch start row
  const float* q = qkv + node * 768 + h * 32;
  float qv[32], acc[32];
#pragma unroll
  for (int d = 0; d < 32; ++d) { qv[d] = q[d]; acc[d] = 0.f; }
  float m = -3.0e38f, l = 0.f;
  const float scale = 0.17677669529663687f;       // 1/sqrt(32)
  for (int kk = 0; kk < 256; ++kk) {
    const float* kr = qkv + (base + kk) * 768 + 256 + h * 32;
    float s = 0.f;
#pragma unroll
    for (int d = 0; d < 32; ++d) s += qv[d] * kr[d];
    s *= scale;
    const float mn = fmaxf(m, s);
    const float corr = __expf(m - mn);
    const float p = __expf(s - mn);
    const float* vr = qkv + (base + kk) * 768 + 512 + h * 32;
    l = l * corr + p;
#pragma unroll
    for (int d = 0; d < 32; ++d) acc[d] = acc[d] * corr + p * vr[d];
    m = mn;
  }
  const float inv = 1.f / l;
  float* op = o + node * 256 + h * 32;
#pragma unroll
  for (int d = 0; d < 32; ++d) op[d] = acc[d] * inv;
}

// ===== input embeddings =====
__global__ __launch_bounds__(256)
void embed_nodes(const float* __restrict__ x, const float* __restrict__ pe,
                 const float* __restrict__ new_w, const float* __restrict__ new_b,
                 const float* __restrict__ png, const float* __restrict__ pnb,
                 const float* __restrict__ plw, const float* __restrict__ plb,
                 float* __restrict__ h)
{
  const long i = blockIdx.x; const int c = threadIdx.x;
  float s;
  if (c < 231) {
    s = new_b[c];
    const float* xp = x + i * IND;
    const float* wp = new_w + (long)c * IND;
    for (int j = 0; j < IND; ++j) s += xp[j] * wp[j];
  } else {
    const int cc = c - 231;
    const float* pp = pe + i * PED;
    float mu = 0.f;
    for (int j = 0; j < PED; ++j) mu += pp[j];
    mu *= (1.f / PED);
    float var = 0.f;
    for (int j = 0; j < PED; ++j) { float d = pp[j] - mu; var += d * d; }
    var *= (1.f / PED);
    const float rs = rsqrtf(var + 1e-5f);
    s = plb[cc];
    const float* wp = plw + (long)cc * PED;
    for (int j = 0; j < PED; ++j)
      s += ((pp[j] - mu) * rs * png[j] + pnb[j]) * wp[j];
  }
  h[i * 256 + c] = s;
}

__global__ __launch_bounds__(256)
void embed_edges(const float* __restrict__ ea_in, const float* __restrict__ w,
                 const float* __restrict__ b, float* __restrict__ ea)
{
  const long e = blockIdx.x; const int c = threadIdx.x;
  const float a0 = ea_in[e * 2], a1 = ea_in[e * 2 + 1];
  ea[e * 256 + c] = a0 * w[c * 2] + a1 * w[c * 2 + 1] + b[c];
}

// ===== final head (64 -> 2) =====
__global__ __launch_bounds__(256)
void head3(const float* __restrict__ in, const float* __restrict__ w,
           const float* __restrict__ b, float* __restrict__ out)
{
  const long i = (long)blockIdx.x * 256 + threadIdx.x;
  const float* p = in + i * 64;
  float o0 = b[0], o1 = b[1];
#pragma unroll
  for (int j = 0; j < 64; ++j) { o0 += p[j] * w[j]; o1 += p[j] * w[64 + j]; }
  out[i * 2] = o0; out[i * 2 + 1] = o1;
}

// =====================================================================
extern "C" void kernel_launch(void* const* d_in, const int* in_sizes, int n_in,
                              void* d_out, int out_size, void* d_ws, size_t ws_size,
                              hipStream_t stream) {
  (void)in_sizes; (void)n_in; (void)out_size; (void)ws_size;
  // inputs in setup_inputs() dict order
  const float* x         = (const float*)d_in[0];
  const int*   edge_idx  = (const int*)  d_in[1];
  const float* edge_attr = (const float*)d_in[2];
  const float* pe        = (const float*)d_in[3];
  // d_in[4] = batch (unused: contiguous groups of 256)
  auto F = [&](int i) { return (const float*)d_in[i]; };
  // params: 5..18 top-level, then 4 layers x 30 entries
  const int LBASE = 19, LSTRIDE = 30;
  auto LP = [&](int l, int j) { return (const float*)d_in[LBASE + l * LSTRIDE + j]; };

  // ---- workspace layout (floats) ----
  float* ws = (float*)d_ws;
  const size_t EAF = (size_t)EE * 256;
  float* ea    = ws;
  float* en    = ws + EAF;
  float* etmp  = ws + 2 * EAF;
  float* p0    = ws + 3 * EAF;
  float* h     = p0;                  p0 += (size_t)NN * 256;
  float* xn    = p0;                  p0 += (size_t)NN * 256;
  float* t     = p0;                  p0 += (size_t)NN * 1024;
  float* xs    = p0;                  p0 += (size_t)NN * 256;
  float* aggr  = p0;                  p0 += (size_t)NN * 256;
  float* nb1   = p0;                  p0 += (size_t)NN * 256;
  float* nb2   = p0;                  p0 += (size_t)NN * 256;
  float* h1    = p0;                  p0 += (size_t)NN * 256;
  float* qkv   = p0;                  p0 += (size_t)NN * 768;
  float* attno = p0;                  p0 += (size_t)NN * 256;
  float* nb3   = p0;                  p0 += (size_t)NN * 256;
  float* obuf  = p0;                  p0 += (size_t)NN * 256;
  float* m1    = p0;                  p0 += (size_t)NN * 512;
  float* xs2   = p0;                  p0 += (size_t)NN * 256;
  float* filt  = p0;                  p0 += (size_t)NN * 512;
  float* hd1   = p0;                  p0 += (size_t)NN * 128;
  float* hd2   = p0;                  p0 += (size_t)NN * 64;
  float* mean  = p0;                  p0 += 256;
  float* rstd  = p0;                  p0 += 256;

  auto gemm = [&](const float* A, const float* W, const float* bias,
                  const float* resid, float* C, long M, int K, int Nout, int act) {
    if ((Nout & 127) == 0) {
      dim3 grid(Nout / 128, (unsigned)(M / 128));
      gemm_bf16_wmma<128><<<grid, 256, 0, stream>>>(A, W, bias, resid, C, M, K, Nout, act);
    } else {
      dim3 grid(Nout / 64, (unsigned)(M / 128));
      gemm_bf16_wmma<64><<<grid, 256, 0, stream>>>(A, W, bias, resid, C, M, K, Nout, act);
    }
  };
  const long NC = (long)NN * 256;

  // ---- embeddings ----
  embed_nodes<<<NN, 256, 0, stream>>>(x, pe, F(5), F(6), F(7), F(8), F(9), F(10), h);
  embed_edges<<<EE, 256, 0, stream>>>(edge_attr, F(11), F(12), ea);

  // ---- layers ----
  for (int l = 0; l < LL; ++l) {
    // LN(h), LN(ea)
    ln256<<<NN / 8, 256, 0, stream>>>(h, LP(l, 0), LP(l, 1), xn, NN, 1e-6f);
    ln256<<<EE / 8, 256, 0, stream>>>(ea, LP(l, 2), LP(l, 3), en, EE, 1e-6f);
    // t = xn @ fcn1^T  (N x 1024)
    gemm(xn, LP(l, 4), LP(l, 5), nullptr, t, NN, 256, 1024, 0);
    slice_xs<<<NC / 256, 256, 0, stream>>>(t, xs);
    // es = en @ fce1[3CH:, :]^T   (only slice ever used)
    gemm(en, LP(l, 8) + 3 * 256 * 256, LP(l, 9) + 3 * 256, nullptr, etmp, EE, 256, 256, 0);
    // ealin = es @ gine_lin^T  -> reuse en
    gemm(etmp, LP(l, 10), LP(l, 11), nullptr, en, EE, 256, 256, 0);
    // aggr = segment_sum(relu(xs[src] + ealin))
    hipMemsetAsync(aggr, 0, (size_t)NN * 256 * sizeof(float), stream);
    gine_msg<<<EE, 256, 0, stream>>>(xs, en, edge_idx, aggr);
    ew_add<<<NC / 256, 256, 0, stream>>>(xs, aggr, aggr);            // h = xs + aggr
    gemm(aggr, LP(l, 12), LP(l, 13), nullptr, nb1, NN, 256, 256, 1); // relu(nn1)
    gemm(nb1, LP(l, 14), LP(l, 15), nullptr, nb2, NN, 256, 256, 0);  // nn2
    bn_stats<<<256, 256, 0, stream>>>(nb2, xs, mean, rstd, NN);
    bn_apply<<<NC / 256, 256, 0, stream>>>(nb2, xs, LP(l, 16), LP(l, 17), mean, rstd, h1);
    // attention branch on xs
    gemm(xs, LP(l, 22), LP(l, 23), nullptr, qkv, NN, 256, 768, 0);
    attn_online<<<(NN * HEADS_) / 256, 256, 0, stream>>>(qkv, attno);
    gemm(attno, LP(l, 24), LP(l, 25), nullptr, nb3, NN, 256, 256, 0);
    bn_stats<<<256, 256, 0, stream>>>(nb3, xs, mean, rstd, NN);
    bn_apply<<<NC / 256, 256, 0, stream>>>(nb3, xs, LP(l, 18), LP(l, 19), mean, rstd, nb2);
    // out = h1 + h2 ; out += mlp(out) ; bn3
    ew_add<<<NC / 256, 256, 0, stream>>>(h1, nb2, obuf);
    gemm(obuf, LP(l, 26), LP(l, 27), nullptr, m1, NN, 256, 512, 1);
    gemm(m1, LP(l, 28), LP(l, 29), obuf, nb1, NN, 512, 256, 0);
    bn_stats<<<256, 256, 0, stream>>>(nb1, nullptr, mean, rstd, NN);
    bn_apply<<<NC / 256, 256, 0, stream>>>(nb1, nullptr, LP(l, 20), LP(l, 21), mean, rstd, xs2);
    // gated output + residual
    gate_gelu<<<((long)NN * 512) / 256, 256, 0, stream>>>(t, xs2, filt);
    gemm(filt, LP(l, 6), LP(l, 7), h, h, NN, 512, 256, 0);           // h = fcn2(filt) + h
  }

  // ---- head ----
  gemm(h,   F(13), F(14), nullptr, hd1, NN, 256, 128, 1);
  gemm(hd1, F(15), F(16), nullptr, hd2, NN, 128, 64, 1);
  head3<<<NN / 256, 256, 0, stream>>>(hd2, F(17), F(18), (float*)d_out);
}
